// SPAttention_89215060672748
// MI455X (gfx1250) — compile-verified, matrix operand order checked by reading, not run on hardware
//
#include <hip/hip_runtime.h>
#include <hip/hip_bf16.h>

// ---------------------------------------------------------------------------
// Fused attention block for MI455X (gfx1250, wave32).
//   ~100 GFLOP vs ~50MB unique bytes -> compute bound -> everything on the
//   WMMA pipe (v_wmma_f32_16x16x32_f16, fp32 accumulate).
//   GEMM staging uses the Tensor Data Mover (tensor_load_to_lds +
//   s_wait_tensorcnt) in a double-buffered pipeline; the TDM pad feature
//   (16-DWORD interval / 4-DWORD pad) produces stride-40-half LDS rows, the
//   bank-conflict-free layout the WMMA fragment loader consumes.
//   Attention reads K/V f16 fragments straight from global: all f16
//   intermediates (~32MB) are resident in the 192MB L2.
// ---------------------------------------------------------------------------

#define T_SEQ 4096
#define C_HID 1024
#define NQKV 3072
#define N_HEADS 16
#define HEAD_D 64
#define NEG_BIG (-3.0e38f)
#define KSTEP 32
#define LDS_STRIDE 40  // 32 halfs + TDM pad (4 DWORDs per 16-DWORD row)

typedef __attribute__((ext_vector_type(16))) _Float16 v16h;
typedef __attribute__((ext_vector_type(8)))  _Float16 v8h;
typedef __attribute__((ext_vector_type(8)))  float    v8f;
typedef __attribute__((ext_vector_type(4)))  unsigned int v4u;
typedef __attribute__((ext_vector_type(8)))  int v8i;
typedef __attribute__((ext_vector_type(4)))  int v4i;

// D = A(16x32 f16) * B(32x16 f16) + C(16x16 f32)
__device__ __forceinline__ v8f wmma32(v16h a, v16h b, v8f c) {
    return __builtin_amdgcn_wmma_f32_16x16x32_f16(
        false, a, false, b, (short)0, c, false, false);
}

// 16x32 f16 fragment (A layout; also B when stored transposed [N][K]).
// lanes 0-15 : row = lane,    K {0..7} and {16..23}
// lanes 16-31: row = lane-16, K {8..15} and {24..31}
__device__ __forceinline__ v16h load_frag(const _Float16* p, int stride) {
    const int lane = threadIdx.x & 31;
    const _Float16* base = p + (lane & 15) * stride + (lane >> 4) * 8;
    v8h lo = *(const v8h*)(base);
    v8h hi = *(const v8h*)(base + 16);
    v16h f;
#pragma unroll
    for (int i = 0; i < 8; ++i) { f[i] = lo[i]; f[i + 8] = hi[i]; }
    return f;
}

__device__ __forceinline__ float rowmax16(float v) {
#pragma unroll
    for (int m = 1; m < 16; m <<= 1) v = fmaxf(v, __shfl_xor(v, m, 32));
    return v;
}
__device__ __forceinline__ float rowsum16(float v) {
#pragma unroll
    for (int m = 1; m < 16; m <<= 1) v += __shfl_xor(v, m, 32);
    return v;
}

// ---------------------------------------------------------------------------
// TDM: issue a 2D f16 tile load (tile = tile_rows x KSTEP elements) from a
// row-major [rows][tensor_k] f16 tensor into LDS, padding each 64B row chunk
// with 16B so LDS rows land at LDS_STRIDE(40) halfs.
// D# layout per CDNA5 ISA; groups 2/3 zero (2D).
// This toolchain's builtin arity is 6: (g0, g1, g2, g3, g4, cpol).
// ---------------------------------------------------------------------------
__device__ __forceinline__ void tdm_issue(const _Float16* gtile,
                                          unsigned lds_off, int tensor_k,
                                          int tensor_rows, int tile_rows) {
    const unsigned long long ga = (unsigned long long)(uintptr_t)gtile;
    v4u g0 = {1u,                                   // count=1 (valid D#)
              lds_off,                              // lds_addr (bytes)
              (unsigned)ga,                         // global_addr[31:0]
              (unsigned)((ga >> 32) & 0x01FFFFFFu)  // global_addr[56:32]
                  | (2u << 30)};                    // type=2 ("image")
    v8i g1;
    g1[0] = (1 << 16)      // data_size = 2 bytes (f16)
          | (1 << 20)      // pad_enable
          | (3 << 22)      // pad_interval: 16 DWORDs
          | (3 << 25);     // pad_amount: 4 DWORDs
    g1[1] = (int)((unsigned)(tensor_k & 0xFFFF) << 16);           // dim0[15:0]
    g1[2] = (int)(((unsigned)tensor_k >> 16) |
                  ((unsigned)(tensor_rows & 0xFFFF) << 16));      // dim0hi|dim1lo
    g1[3] = (int)(((unsigned)tensor_rows >> 16) | (KSTEP << 16)); // dim1hi|tile0
    g1[4] = tile_rows;   // tile_dim1 (tile_dim2 = 0)
    g1[5] = tensor_k;    // tensor_dim0_stride[31:0]
    g1[6] = 0;
    g1[7] = 0;
    const v4i z4 = {0, 0, 0, 0};
    const v8i z8 = {0, 0, 0, 0, 0, 0, 0, 0};
    __builtin_amdgcn_tensor_load_to_lds(g0, g1, z4, z4, z8, 0);
}

// ---------------------------------------------------------------------------
// Pre-pass 1: fp32 -> f16 copy (for x).
// ---------------------------------------------------------------------------
__global__ __launch_bounds__(256) void cvt_f16_kernel(
    const float* __restrict__ src, _Float16* __restrict__ dst, int n) {
    const int i = (blockIdx.x * 256 + threadIdx.x) * 4;
    if (i < n) {
        const float4 v = *(const float4*)(src + i);
        dst[i + 0] = (_Float16)v.x;
        dst[i + 1] = (_Float16)v.y;
        dst[i + 2] = (_Float16)v.z;
        dst[i + 3] = (_Float16)v.w;
    }
}

// ---------------------------------------------------------------------------
// Pre-pass 2: fp32 [K][N] -> f16 [N][K] transpose (weights), 32x32 LDS tiles.
// ---------------------------------------------------------------------------
__global__ __launch_bounds__(256) void transpose_f16_kernel(
    const float* __restrict__ src, _Float16* __restrict__ dst, int K, int N) {
    __shared__ float tile[32][33];
    const int n0 = blockIdx.x * 32;
    const int k0 = blockIdx.y * 32;
    const int c = threadIdx.x & 31;
    const int r4 = threadIdx.x >> 5;  // 0..7
#pragma unroll
    for (int i = 0; i < 4; ++i) {
        const int r = r4 * 4 + i;
        tile[r][c] = src[(size_t)(k0 + r) * N + n0 + c];
    }
    __syncthreads();
#pragma unroll
    for (int i = 0; i < 4; ++i) {
        const int r = r4 * 4 + i;
        dst[(size_t)(n0 + r) * K + k0 + c] = (_Float16)tile[c][r];
    }
}

// ---------------------------------------------------------------------------
// Shared GEMM mainloop: acc(32x64 per wave) over K=1024 in steps of 32 with
// double-buffered TDM staging. Block tile 128(M) x 128(N), 8 waves (4m x 2n).
// A: [T_SEQ][1024] f16, Bt: [rowsB][1024] f16 (transposed weights).
// ---------------------------------------------------------------------------
__device__ __forceinline__ void gemm_mainloop(const _Float16* __restrict__ A,
                                              const _Float16* __restrict__ Bt,
                                              int m0, int n0, int rowsB,
                                              _Float16* ldsA, _Float16* ldsB,
                                              v8f acc[2][4]) {
    const int wave = threadIdx.x >> 5;
    const int wm = wave & 3;   // 0..3 -> 32-row slice
    const int wn = wave >> 2;  // 0..1 -> 64-col slice
    const int NIT = C_HID / KSTEP;
    const unsigned BUFB = 128 * LDS_STRIDE * sizeof(_Float16);  // bytes/buffer

    const _Float16* Abase = A + (size_t)m0 * C_HID;
    const _Float16* Bbase = Bt + (size_t)n0 * C_HID;
    const unsigned ldsAoff = (unsigned)(uintptr_t)ldsA;
    const unsigned ldsBoff = (unsigned)(uintptr_t)ldsB;

    if (wave == 0) {  // prologue: tile 0 -> buffer 0
        tdm_issue(Abase, ldsAoff, C_HID, T_SEQ, 128);
        tdm_issue(Bbase, ldsBoff, C_HID, rowsB, 128);
    }
    for (int it = 0; it < NIT; ++it) {
        const int buf = it & 1;
        if (wave == 0) {
            if (it + 1 < NIT) {  // issue next tile into other buffer
                const int kn = (it + 1) * KSTEP;
                tdm_issue(Abase + kn, ldsAoff + (buf ^ 1) * BUFB, C_HID, T_SEQ,
                          128);
                tdm_issue(Bbase + kn, ldsBoff + (buf ^ 1) * BUFB, C_HID, rowsB,
                          128);
                __builtin_amdgcn_s_wait_tensorcnt(2);  // tile `it` complete
            } else {
                __builtin_amdgcn_s_wait_tensorcnt(0);
            }
        }
        __syncthreads();  // tile `it` visible to all waves
        const _Float16* pa = ldsA + buf * (128 * LDS_STRIDE) +
                             (wm * 32) * LDS_STRIDE;
        const _Float16* pb = ldsB + buf * (128 * LDS_STRIDE) +
                             (wn * 64) * LDS_STRIDE;
        const v16h a0 = load_frag(pa, LDS_STRIDE);
        const v16h a1 = load_frag(pa + 16 * LDS_STRIDE, LDS_STRIDE);
#pragma unroll
        for (int j = 0; j < 4; ++j) {
            const v16h bf = load_frag(pb + j * 16 * LDS_STRIDE, LDS_STRIDE);
            acc[0][j] = wmma32(a0, bf, acc[0][j]);
            acc[1][j] = wmma32(a1, bf, acc[1][j]);
        }
        __syncthreads();  // reads done before buffer is re-filled
    }
}

// ---------------------------------------------------------------------------
// Kernel 1: QKV projection -> Q[h][t][d], K[h][t][d], Vt[h][d][t] (f16).
// ---------------------------------------------------------------------------
__global__ __launch_bounds__(256) void qkv_gemm_kernel(
    const _Float16* __restrict__ xh, const _Float16* __restrict__ wT,
    const float* __restrict__ b, _Float16* __restrict__ qws,
    _Float16* __restrict__ kws, _Float16* __restrict__ vtws) {
    __shared__ _Float16 ldsA[2 * 128 * LDS_STRIDE];
    __shared__ _Float16 ldsB[2 * 128 * LDS_STRIDE];
    const int lane = threadIdx.x & 31;
    const int wave = threadIdx.x >> 5;
    const int wm = wave & 3, wn = wave >> 2;
    const int m0 = blockIdx.y * 128, n0 = blockIdx.x * 128;

    v8f acc[2][4] = {};
    gemm_mainloop(xh, wT, m0, n0, NQKV, ldsA, ldsB, acc);

    const int lanecol = lane & 15, hi = lane >> 4;
#pragma unroll
    for (int fi = 0; fi < 2; ++fi)
#pragma unroll
        for (int fj = 0; fj < 4; ++fj)
#pragma unroll
            for (int r = 0; r < 8; ++r) {
                const int grow = m0 + wm * 32 + fi * 16 + r + 8 * hi;
                const int gcol = n0 + wn * 64 + fj * 16 + lanecol;
                const float v = acc[fi][fj][r] + b[gcol];
                const int part = gcol >> 10;  // 0=q 1=k 2=v
                const int rem = gcol & 1023;
                const int h = rem >> 6;
                const int d = rem & 63;
                if (part == 0)
                    qws[((size_t)h * T_SEQ + grow) * HEAD_D + d] = (_Float16)v;
                else if (part == 1)
                    kws[((size_t)h * T_SEQ + grow) * HEAD_D + d] = (_Float16)v;
                else
                    vtws[((size_t)h * HEAD_D + d) * T_SEQ + grow] = (_Float16)v;
            }
}

// ---------------------------------------------------------------------------
// Kernel 2: causal flash attention. 4 waves/block, 16 query rows per wave,
// key blocks of 32. K/V fragments straight from global (L2-resident).
// ---------------------------------------------------------------------------
__global__ __launch_bounds__(128) void attn_kernel(
    const _Float16* __restrict__ qws, const _Float16* __restrict__ kws,
    const _Float16* __restrict__ vtws, _Float16* __restrict__ ows) {
    __shared__ _Float16 sP[4 * 16 * LDS_STRIDE];  // per-wave exp(P) staging

    const int wave = threadIdx.x >> 5;
    const int lane = threadIdx.x & 31;
    const int lanecol = lane & 15;
    const int hi = lane >> 4;
    const int head = blockIdx.x >> 6;  // 64 q-blocks of 64 rows
    const int qblk = blockIdx.x & 63;
    const int qb = qblk * 64 + wave * 16;

    const _Float16* qbase = qws + ((size_t)head * T_SEQ + qb) * HEAD_D;
    const _Float16* kbase = kws + (size_t)head * T_SEQ * HEAD_D;
    const _Float16* vtb = vtws + (size_t)head * HEAD_D * T_SEQ;
    _Float16* sp = sP + wave * 16 * LDS_STRIDE;

    const v16h qf0 = load_frag(qbase, HEAD_D);       // dims 0..31
    const v16h qf1 = load_frag(qbase + 32, HEAD_D);  // dims 32..63

    v8f o[4] = {};
    float rmax[8], rsum[8];
#pragma unroll
    for (int r = 0; r < 8; ++r) { rmax[r] = NEG_BIG; rsum[r] = 0.0f; }

    for (int k0 = 0; k0 < qb + 16; k0 += 32) {
        // ---- S = scale * Q @ K^T, two 16-key tiles ----
        v8f s0 = {}, s1 = {};
        s0 = wmma32(qf0, load_frag(kbase + (size_t)(k0 + 0) * HEAD_D + 0, HEAD_D), s0);
        s0 = wmma32(qf1, load_frag(kbase + (size_t)(k0 + 0) * HEAD_D + 32, HEAD_D), s0);
        s1 = wmma32(qf0, load_frag(kbase + (size_t)(k0 + 16) * HEAD_D + 0, HEAD_D), s1);
        s1 = wmma32(qf1, load_frag(kbase + (size_t)(k0 + 16) * HEAD_D + 32, HEAD_D), s1);

        // ---- causal mask + online softmax ----
#pragma unroll
        for (int r = 0; r < 8; ++r) {
            const int q = qb + r + 8 * hi;
            float a = s0[r] * 0.125f;  // 1/sqrt(64)
            float c = s1[r] * 0.125f;
            a = ((k0 + lanecol) <= q) ? a : NEG_BIG;
            c = ((k0 + 16 + lanecol) <= q) ? c : NEG_BIG;
            const float stepm = rowmax16(fmaxf(a, c));
            const float nm = fmaxf(rmax[r], stepm);
            const float alpha = __expf(rmax[r] - nm);
            rmax[r] = nm;
            const float p0 = __expf(a - nm);
            const float p1 = __expf(c - nm);
            rsum[r] = rsum[r] * alpha + rowsum16(p0 + p1);
            o[0][r] *= alpha; o[1][r] *= alpha;
            o[2][r] *= alpha; o[3][r] *= alpha;
            const int row = r + 8 * hi;
            sp[row * LDS_STRIDE + lanecol] = (_Float16)p0;
            sp[row * LDS_STRIDE + lanecol + 16] = (_Float16)p1;
        }
        // cross-lane LDS visibility within the wave (no barrier: waves have
        // divergent trip counts; DS ops are in-order per wave)
        asm volatile("s_wait_dscnt 0" ::: "memory");

        // ---- O += P @ V (V^T [dim][seq] is the natural B layout) ----
        const v16h pf = load_frag(sp, LDS_STRIDE);
#pragma unroll
        for (int c = 0; c < 4; ++c)
            o[c] = wmma32(
                pf, load_frag(vtb + (size_t)(c * 16) * T_SEQ + k0, T_SEQ), o[c]);
    }

#pragma unroll
    for (int c = 0; c < 4; ++c)
#pragma unroll
        for (int r = 0; r < 8; ++r) {
            const int row = qb + r + 8 * hi;
            const int col = head * HEAD_D + c * 16 + lanecol;
            ows[(size_t)row * C_HID + col] = (_Float16)(o[c][r] / rsum[r]);
        }
}

// ---------------------------------------------------------------------------
// Kernel 3: output projection. out[4096,1024] fp32 = attn @ w_out + b_out.
// ---------------------------------------------------------------------------
__global__ __launch_bounds__(256) void out_gemm_kernel(
    const _Float16* __restrict__ aws, const _Float16* __restrict__ wT,
    const float* __restrict__ b, float* __restrict__ out) {
    __shared__ _Float16 ldsA[2 * 128 * LDS_STRIDE];
    __shared__ _Float16 ldsB[2 * 128 * LDS_STRIDE];
    const int lane = threadIdx.x & 31;
    const int wave = threadIdx.x >> 5;
    const int wm = wave & 3, wn = wave >> 2;
    const int m0 = blockIdx.y * 128, n0 = blockIdx.x * 128;

    v8f acc[2][4] = {};
    gemm_mainloop(aws, wT, m0, n0, C_HID, ldsA, ldsB, acc);

    const int lanecol = lane & 15, hi = lane >> 4;
#pragma unroll
    for (int fi = 0; fi < 2; ++fi)
#pragma unroll
        for (int fj = 0; fj < 4; ++fj)
#pragma unroll
            for (int r = 0; r < 8; ++r) {
                const int grow = m0 + wm * 32 + fi * 16 + r + 8 * hi;
                const int gcol = n0 + wn * 64 + fj * 16 + lanecol;
                out[(size_t)grow * C_HID + gcol] = acc[fi][fj][r] + b[gcol];
            }
}

// ---------------------------------------------------------------------------
extern "C" void kernel_launch(void* const* d_in, const int* in_sizes, int n_in,
                              void* d_out, int out_size, void* d_ws,
                              size_t ws_size, hipStream_t stream) {
    (void)in_sizes; (void)n_in; (void)out_size; (void)ws_size;
    const float* x = (const float*)d_in[0];      // [1,4096,1024]
    const float* w_qkv = (const float*)d_in[1];  // [1024,3072]
    const float* b_qkv = (const float*)d_in[2];  // [3072]
    const float* w_out = (const float*)d_in[3];  // [1024,1024]
    const float* b_out = (const float*)d_in[4];  // [1024]
    float* out = (float*)d_out;                  // [4096,1024]

    const size_t HTD = (size_t)N_HEADS * T_SEQ * HEAD_D;  // 4 Mi halfs
    _Float16* qws = (_Float16*)d_ws;           // [h][t][d]
    _Float16* kws = qws + HTD;                 // [h][t][d]
    _Float16* vtws = kws + HTD;                // [h][d][t]
    _Float16* aws = vtws + HTD;                // [t][c] attention output
    _Float16* xh = aws + (size_t)T_SEQ * C_HID;        // [t][c] f16 x
    _Float16* wqkvT = xh + (size_t)T_SEQ * C_HID;      // [3072][1024]
    _Float16* woutT = wqkvT + (size_t)NQKV * C_HID;    // [1024][1024]

    // one-shot precision/layout pre-pass (trivial vs 23.3 TB/s)
    cvt_f16_kernel<<<dim3((T_SEQ * C_HID) / 1024), 256, 0, stream>>>(
        x, xh, T_SEQ * C_HID);
    transpose_f16_kernel<<<dim3(NQKV / 32, C_HID / 32), 256, 0, stream>>>(
        w_qkv, wqkvT, C_HID, NQKV);
    transpose_f16_kernel<<<dim3(C_HID / 32, C_HID / 32), 256, 0, stream>>>(
        w_out, woutT, C_HID, C_HID);

    qkv_gemm_kernel<<<dim3(NQKV / 128, T_SEQ / 128), 256, 0, stream>>>(
        xh, wqkvT, b_qkv, qws, kws, vtws);
    attn_kernel<<<dim3(N_HEADS * (T_SEQ / 64)), 128, 0, stream>>>(
        qws, kws, vtws, aws);
    out_gemm_kernel<<<dim3(C_HID / 128, T_SEQ / 128), 256, 0, stream>>>(
        aws, woutT, b_out, out);
}
